// HybridMemorySystem_71717363909231
// MI455X (gfx1250) — matrix operand order repaired and checked by reading, not compiled, not run on hardware
//
#include <hip/hip_runtime.h>
#include <hip/hip_bf16.h>

#define EPS 1e-5f
#define ALPHA 0.1f
#define MSPLIT 4                       // M-dimension partitions for knn kernel

typedef _Float16 v16h __attribute__((ext_vector_type(16)));
typedef _Float16 v8h  __attribute__((ext_vector_type(8)));
typedef float    v8f  __attribute__((ext_vector_type(8)));
typedef float    v4f  __attribute__((ext_vector_type(4)));

// ---------------------------------------------------------------------------
// Sorted ascending top-16 insertion (all static indexing after unroll; caller
// guarantees x < tv[15]).
// ---------------------------------------------------------------------------
__device__ __forceinline__ void topk_insert16(float tv[16], int ti[16], float x, int xi) {
    float cv = x; int ci = xi;
    bool done = false;
#pragma unroll
    for (int i = 15; i >= 1; --i) {
        float pb = tv[i - 1]; int pj = ti[i - 1];
        bool shift = (!done) && (cv < pb);
        if (shift)      { tv[i] = pb; ti[i] = pj; }
        else if (!done) { tv[i] = cv; ti[i] = ci; done = true; }
    }
    if (!done) { tv[0] = cv; ti[0] = ci; }
}

// ---------------------------------------------------------------------------
// Kernel 1: query network. One block per query row (Nq=2048), 256 threads.
// q = gelu(LN(hidden@w1+b1)) @ w2 + b2 ; poincare project ; emit f16 row + |q|^2
// ---------------------------------------------------------------------------
__global__ __launch_bounds__(256) void hms_query_net(
    const float* __restrict__ hidden, const float* __restrict__ w1,
    const float* __restrict__ b1, const float* __restrict__ ln_g,
    const float* __restrict__ ln_b, const float* __restrict__ w2,
    const float* __restrict__ b2, _Float16* __restrict__ qh,
    float* __restrict__ x2q)
{
    __shared__ float sh[1024];
    __shared__ float red[256];
    __shared__ float sg[256];
    const int n = blockIdx.x;
    const int d = threadIdx.x;

    for (int i = d; i < 1024; i += 256) sh[i] = hidden[(size_t)n * 1024 + i];
    __syncthreads();

    float acc = b1[d];
    for (int h = 0; h < 1024; ++h) acc = fmaf(sh[h], w1[(size_t)h * 256 + d], acc);

    // mean
    red[d] = acc; __syncthreads();
    for (int s = 128; s > 0; s >>= 1) { if (d < s) red[d] += red[d + s]; __syncthreads(); }
    float mu = red[0] * (1.0f / 256.0f);
    __syncthreads();
    // variance (biased, ddof=0)
    float cm = acc - mu;
    red[d] = cm * cm; __syncthreads();
    for (int s = 128; s > 0; s >>= 1) { if (d < s) red[d] += red[d + s]; __syncthreads(); }
    float var = red[0] * (1.0f / 256.0f);
    __syncthreads();

    float y = cm * rsqrtf(var + EPS) * ln_g[d] + ln_b[d];
    float g = 0.5f * y * (1.0f + erff(y * 0.70710678118654752f));   // exact GELU
    sg[d] = g; __syncthreads();

    float q2 = b2[d];
    for (int k = 0; k < 256; ++k) q2 = fmaf(sg[k], w2[(size_t)k * 256 + d], q2);

    // poincare projection (c = 1)
    red[d] = q2 * q2; __syncthreads();
    for (int s = 128; s > 0; s >>= 1) { if (d < s) red[d] += red[d + s]; __syncthreads(); }
    float n2 = red[0];
    float nrm = sqrtf(n2);
    float maxn = 1.0f - EPS;
    float scale = (nrm > maxn) ? (maxn / fmaxf(nrm, EPS)) : 1.0f;
    qh[(size_t)n * 256 + d] = (_Float16)(q2 * scale);
    if (d == 0) x2q[n] = n2 * scale * scale;
}

// ---------------------------------------------------------------------------
// Kernel 2: project memory bank onto the ball; emit f16 rows, |m|^2, scale.
// One block per memory row (M=65536), 256 threads.
// ---------------------------------------------------------------------------
__global__ __launch_bounds__(256) void hms_mem_proj(
    const float* __restrict__ memory, _Float16* __restrict__ mh,
    float* __restrict__ y2m, float* __restrict__ scale_m)
{
    __shared__ float red[256];
    const int m = blockIdx.x;
    const int d = threadIdx.x;
    float v = memory[(size_t)m * 256 + d];
    red[d] = v * v; __syncthreads();
    for (int s = 128; s > 0; s >>= 1) { if (d < s) red[d] += red[d + s]; __syncthreads(); }
    float n2 = red[0];
    float nrm = sqrtf(n2);
    float maxn = 1.0f - EPS;
    float scale = (nrm > maxn) ? (maxn / fmaxf(nrm, EPS)) : 1.0f;
    mh[(size_t)m * 256 + d] = (_Float16)(v * scale);
    if (d == 0) { y2m[m] = n2 * scale * scale; scale_m[m] = scale; }
}

// ---------------------------------------------------------------------------
// Kernel 3: WMMA Poincare distances + fused per-lane top-16.
// Grid: (128, MSPLIT) blocks; block = 16 queries x one M-partition, 8 waves.
// A = 16 memory rows (M), B = 16 query cols (N), K = 256 -> 8x wmma f16 K=32.
// C-layout: lane holds 8 dists for ONE query (N = lane%16) -> register top-k.
// Software pipeline at half-tile (K=128) granularity; static buffers a0/a1;
// the t+1 preload is UNCONDITIONAL (clamped pointer on the last iteration) so
// there is no control-flow merge and every s_wait_loadcnt stays precise.
// Rank by monotone surrogate arg (v_rcp); acosh deferred to the merge kernel.
// ---------------------------------------------------------------------------
__device__ __forceinline__ void hms_load_half(v16h af[4], const _Float16* __restrict__ p) {
#pragma unroll
    for (int kk = 0; kk < 4; ++kk) {
        v8h lo = *(const v8h*)(p + kk * 32);
        v8h hv = *(const v8h*)(p + kk * 32 + 8);
        af[kk] = __builtin_shufflevector(lo, hv, 0, 1, 2, 3, 4, 5, 6, 7,
                                                 8, 9, 10, 11, 12, 13, 14, 15);
    }
}

__global__ __launch_bounds__(256) void hms_knn_topk(
    const _Float16* __restrict__ qh, const float* __restrict__ x2q,
    const _Float16* __restrict__ mh, const float* __restrict__ y2m,
    float* __restrict__ topka, int* __restrict__ topkai)
{
    __shared__ float sdv[256 * 16];
    __shared__ int   sdi[256 * 16];

    const int tid   = threadIdx.x;
    const int wave  = tid >> 5;
    const int lane  = tid & 31;
    const int col   = lane & 15;   // query column within tile
    const int hi    = lane >> 4;   // lane half selects K sub-block / M sub-block
    const int qbase = blockIdx.x * 16;
    const int split = blockIdx.y;  // M partition

    // B fragments (queries), held in registers for the whole M loop.
    // B k order chosen here (matches A): lanes 0-15 hold K=0..15 of column N=lane, lanes 16-31 K=16..31.
    v16h bfrag[8];
    {
        const _Float16* qrow = qh + (size_t)(qbase + col) * 256 + hi * 16;
#pragma unroll
        for (int kk = 0; kk < 8; ++kk)
            bfrag[kk] = *(const v16h*)(qrow + kk * 32);
    }
    const float x2j   = x2q[qbase + col];
    const float onemx = 1.0f - x2j;

    float tv[16]; int ti[16];
#pragma unroll
    for (int i = 0; i < 16; ++i) { tv[i] = 3.4e38f; ti[i] = 0; }

    const int NT    = 65536 / MSPLIT / 8 / 16;              // 128 tiles per wave
    const int wbase = split * (65536 / MSPLIT) + wave * (NT * 16);

    // A k order chosen here (matches B): lane holds row M=lane%16; lanes<16 K 0..15,
    // lanes>=16 K 16..31 -> two adjacent 16B chunks per k-step.
    const _Float16* ap = mh + (size_t)(wbase + col) * 256 + hi * 16;
    const float*    yp = y2m + wbase + hi * 8;  // 8 consecutive y2 per lane-half

    v16h a0[4], a1[4];                          // static half-tile buffers
    hms_load_half(a0, ap);                      // K-lo half of tile 0

    for (int t = 0; t < NT; ++t) {
        const int mbase = wbase + t * 16;

        hms_load_half(a1, ap + 128);            // K-hi half of tile t
        v4f ya = *(const v4f*)(yp);
        v4f yc = *(const v4f*)(yp + 4);
        __builtin_prefetch((const void*)(ap + 2 * 16 * 256), 0, 1);  // t+2 hint

        v8f acc = {0.f, 0.f, 0.f, 0.f, 0.f, 0.f, 0.f, 0.f};
#pragma unroll
        for (int kk = 0; kk < 4; ++kk)
            acc = __builtin_amdgcn_wmma_f32_16x16x32_f16(
                false, a0[kk], false, bfrag[kk], (short)0, acc, false, false);

        // unconditional preload of tile t+1's K-lo half (clamped on last iter)
        const _Float16* apn = ap + 16 * 256;
        const _Float16* pl  = (t < NT - 1) ? apn : ap;
        hms_load_half(a0, pl);
        ap = apn;
        yp += 16;

#pragma unroll
        for (int kk = 0; kk < 4; ++kk)
            acc = __builtin_amdgcn_wmma_f32_16x16x32_f16(
                false, a1[kk], false, bfrag[kk + 4], (short)0, acc, false, false);

        // C layout: VGPR v -> memory row mbase + v + 8*hi, column = query col.
#pragma unroll
        for (int v = 0; v < 8; ++v) {
            const int   m  = mbase + hi * 8 + v;
            const float y2 = (v < 4) ? ya[v & 3] : yc[v & 3];
            float sq    = fmaxf(x2j + y2 - 2.0f * acc[v], 0.0f);
            float denom = fmaxf(onemx * (1.0f - y2), EPS);
            float arg   = fmaxf(fmaf(2.0f * sq, __builtin_amdgcn_rcpf(denom), 1.0f),
                                1.0f + EPS);
            if (arg < tv[15]) topk_insert16(tv, ti, arg, m);
        }
    }

    // dump per-lane lists, merge 16 contributing lanes per query
#pragma unroll
    for (int i = 0; i < 16; ++i) { sdv[tid * 16 + i] = tv[i]; sdi[tid * 16 + i] = ti[i]; }
    __syncthreads();

    if (tid < 16) {
        float mv[16]; int mi[16];
#pragma unroll
        for (int i = 0; i < 16; ++i) { mv[i] = 3.4e38f; mi[i] = 0; }
        for (int w = 0; w < 8; ++w)
            for (int h = 0; h < 2; ++h) {
                int src = w * 32 + h * 16 + tid;
                for (int e = 0; e < 16; ++e) {
                    float x = sdv[src * 16 + e];
                    if (x >= mv[15]) break;     // source list ascending
                    topk_insert16(mv, mi, x, sdi[src * 16 + e]);
                }
            }
        // store surrogate args (ascending) for this M partition
        const size_t base = ((size_t)(qbase + tid) * MSPLIT + split) * 16;
        for (int k = 0; k < 16; ++k) {
            topka[base + k]  = mv[k];
            topkai[base + k] = mi[k];
        }
    }
}

// ---------------------------------------------------------------------------
// Kernel 3b: merge the MSPLIT partial sorted lists per query; evaluate acosh
// only for the final 16 winners. One thread per query.
// ---------------------------------------------------------------------------
__global__ __launch_bounds__(128) void hms_topk_merge(
    const float* __restrict__ topka, const int* __restrict__ topkai,
    float* __restrict__ topkd, int* __restrict__ topki)
{
    const int q = blockIdx.x * 128 + threadIdx.x;
    float mv[16]; int mi[16];
#pragma unroll
    for (int i = 0; i < 16; ++i) { mv[i] = 3.4e38f; mi[i] = 0; }
    for (int s = 0; s < MSPLIT; ++s) {
        const float* pv = topka  + ((size_t)q * MSPLIT + s) * 16;
        const int*   pi = topkai + ((size_t)q * MSPLIT + s) * 16;
        for (int e = 0; e < 16; ++e) {
            float x = pv[e];
            if (x >= mv[15]) break;             // source list ascending
            topk_insert16(mv, mi, x, pi[e]);
        }
    }
    for (int k = 0; k < 16; ++k) {
        float a = mv[k];
        float d = logf(a + sqrtf(fmaxf(a * a - 1.0f, 0.0f)));   // acosh, sqrt(c)=1
        topkd[(size_t)q * 16 + k] = d;
        topki[(size_t)q * 16 + k] = mi[k];
    }
}

// ---------------------------------------------------------------------------
// Kernel 4: softmax(-dist), gather+aggregate retrieved rows, project to hidden,
// residual blend. One block per query row, 256 threads.
// ---------------------------------------------------------------------------
__global__ __launch_bounds__(256) void hms_aggregate(
    const float* __restrict__ hidden, const float* __restrict__ memory,
    const float* __restrict__ scale_m, const float* __restrict__ topkd,
    const int* __restrict__ topki, const float* __restrict__ wp,
    const float* __restrict__ bp, float* __restrict__ out)
{
    __shared__ float sd[16];
    __shared__ int   si[16];
    __shared__ float sr[256];
    const int n = blockIdx.x;
    const int t = threadIdx.x;

    if (t < 16) { sd[t] = topkd[(size_t)n * 16 + t]; si[t] = topki[(size_t)n * 16 + t]; }
    __syncthreads();

    // softmax over -dist (redundant per thread; 16 values)
    float wloc[16];
    float mx = -sd[0];
#pragma unroll
    for (int k = 1; k < 16; ++k) mx = fmaxf(mx, -sd[k]);
    float s = 0.0f;
#pragma unroll
    for (int k = 0; k < 16; ++k) { wloc[k] = expf(-sd[k] - mx); s += wloc[k]; }
    float inv = 1.0f / s;

    // retrieved[d] = sum_k w_k * (memory[idx_k] * proj_scale[idx_k])[d]
    float r = 0.0f;
#pragma unroll
    for (int k = 0; k < 16; ++k) {
        int id = si[k];
        r = fmaf(wloc[k] * inv * scale_m[id], memory[(size_t)id * 256 + t], r);
    }
    sr[t] = r;
    __syncthreads();

    for (int h = t; h < 1024; h += 256) {
        float a = bp[h];
        for (int dd = 0; dd < 256; ++dd) a = fmaf(sr[dd], wp[(size_t)dd * 1024 + h], a);
        out[(size_t)n * 1024 + h] = hidden[(size_t)n * 1024 + h] + ALPHA * a;
    }
}

// ---------------------------------------------------------------------------
// Launch
// ---------------------------------------------------------------------------
extern "C" void kernel_launch(void* const* d_in, const int* in_sizes, int n_in,
                              void* d_out, int out_size, void* d_ws, size_t ws_size,
                              hipStream_t stream) {
    (void)in_sizes; (void)n_in; (void)out_size; (void)ws_size;

    const float* hidden = (const float*)d_in[0];   // [2048,1024]
    const float* memory = (const float*)d_in[1];   // [65536,256]
    const float* w1     = (const float*)d_in[2];   // [1024,256]
    const float* b1     = (const float*)d_in[3];   // [256]
    const float* ln_g   = (const float*)d_in[4];   // [256]
    const float* ln_b   = (const float*)d_in[5];   // [256]
    const float* w2     = (const float*)d_in[6];   // [256,256]
    const float* b2     = (const float*)d_in[7];   // [256]
    const float* wp     = (const float*)d_in[8];   // [256,1024]
    const float* bp     = (const float*)d_in[9];   // [1024]
    float* out = (float*)d_out;                    // [2048,1024]

    // workspace carve-up (256B-aligned sections), ~36.5 MB total
    char* w = (char*)d_ws;
    size_t off = 0;
    _Float16* mh = (_Float16*)(w + off); off += (size_t)65536 * 256 * 2;   // 32 MB
    _Float16* qh = (_Float16*)(w + off); off += (size_t)2048 * 256 * 2;    // 1 MB
    float* x2q     = (float*)(w + off);  off += (size_t)2048 * 4;
    off = (off + 255) & ~(size_t)255;
    float* y2m     = (float*)(w + off);  off += (size_t)65536 * 4;
    float* scale_m = (float*)(w + off);  off += (size_t)65536 * 4;
    float* topka   = (float*)(w + off);  off += (size_t)2048 * MSPLIT * 16 * 4;
    int*   topkai  = (int*)(w + off);    off += (size_t)2048 * MSPLIT * 16 * 4;
    float* topkd   = (float*)(w + off);  off += (size_t)2048 * 16 * 4;
    int*   topki   = (int*)(w + off);    off += (size_t)2048 * 16 * 4;

    hms_query_net<<<2048, 256, 0, stream>>>(hidden, w1, b1, ln_g, ln_b, w2, b2, qh, x2q);
    hms_mem_proj<<<65536, 256, 0, stream>>>(memory, mh, y2m, scale_m);
    hms_knn_topk<<<dim3(128, MSPLIT), 256, 0, stream>>>(qh, x2q, mh, y2m, topka, topkai);
    hms_topk_merge<<<2048 / 128, 128, 0, stream>>>(topka, topkai, topkd, topki);
    hms_aggregate<<<2048, 256, 0, stream>>>(hidden, memory, scale_m, topkd, topki, wp, bp, out);
}
